// SimplifiedCTM_36747740185053
// MI455X (gfx1250) — matrix operand order repaired and checked by reading, not compiled, not run on hardware
//
#include <hip/hip_runtime.h>
#include <hip/hip_bf16.h>

typedef __attribute__((ext_vector_type(16))) _Float16 v16h;
typedef __attribute__((ext_vector_type(8)))  _Float16 v8h;
typedef __attribute__((ext_vector_type(8)))  float    v8f;

#define NN 4096
#define MM 10
#define TT 10
#define BB 1024
#define PP 1024

// ---------- helpers ----------
__device__ __forceinline__ float block_reduce_sum(float val, float* buf) {
  int tid = threadIdx.x;
  int n = blockDim.x;            // power of two
  buf[tid] = val;
  __syncthreads();
  for (int s = n >> 1; s > 0; s >>= 1) {
    if (tid < s) buf[tid] += buf[tid + s];
    __syncthreads();
  }
  float r = buf[0];
  __syncthreads();
  return r;
}

__global__ void copy_f32(const float* __restrict__ src, float* __restrict__ dst, int n) {
  int i = blockIdx.x * blockDim.x + threadIdx.x;
  if (i < n) dst[i] = src[i];
}

// ---------- operand packing: f32 -> zero-padded f16 ----------
// Row-major A: out[r*Kpad + k] = A[r*lda + k]  (k >= K -> 0)
__global__ void cvt_f16_pad(const float* __restrict__ A, int K, int lda, int Kpad,
                            _Float16* __restrict__ out, int total) {
  int idx = blockIdx.x * blockDim.x + threadIdx.x;
  if (idx >= total) return;
  int r = idx / Kpad, k = idx % Kpad;
  out[idx] = (k < K) ? (_Float16)A[r * lda + k] : (_Float16)0.0f;
}

// K-major (transposed) B: out[n*Kpad + k] = B[k*ldb + n]  (k >= K -> 0)
__global__ void cvt_f16_pad_t(const float* __restrict__ B, int K, int ldb, int Kpad,
                              _Float16* __restrict__ out, int total) {
  int idx = blockIdx.x * blockDim.x + threadIdx.x;
  if (idx >= total) return;
  int n = idx / Kpad, k = idx % Kpad;
  out[idx] = (k < K) ? (_Float16)B[k * ldb + n] : (_Float16)0.0f;
}

// ---------- WMMA GEMM on packed f16 operands ----------
// C(f32, ldc) = Ah(M x Kpad, row-major f16) * Bt(N x Kpad, K-major f16)^T
// One wave computes a 16x64 output strip (4 N-tiles), reusing the A fragment.
// wavecols = N/64. Kpad % 32 == 0; all fragment loads contiguous & 16B/32B aligned.
__global__ void wmma_gemm_packed(const _Float16* __restrict__ Ah,
                                 const _Float16* __restrict__ Bt,
                                 float* __restrict__ C, int ldc,
                                 int Kpad, int wavecols) {
  int wave = (blockIdx.x * blockDim.x + threadIdx.x) >> 5;
  int lane = threadIdx.x & 31;
  int tm  = (wave / wavecols) * 16;
  int tn0 = (wave % wavecols) * 64;
  int mr = lane & 15;     // A row / B col / C col within tile
  int hi = lane >> 4;     // K-phase select
  int kbA = hi * 16;      // A: elems 0..15 -> K kbA..kbA+15
  int kbB = hi * 16;      // B: elems 0..15 -> K kbB..kbB+15

  const _Float16* arow = Ah + (size_t)(tm + mr) * Kpad;
  const _Float16* b0 = Bt + (size_t)(tn0 + mr) * Kpad;
  const _Float16* b1 = Bt + (size_t)(tn0 + 16 + mr) * Kpad;
  const _Float16* b2 = Bt + (size_t)(tn0 + 32 + mr) * Kpad;
  const _Float16* b3 = Bt + (size_t)(tn0 + 48 + mr) * Kpad;

  v8f acc0 = {}, acc1 = {}, acc2 = {}, acc3 = {};
  for (int k0 = 0; k0 < Kpad; k0 += 32) {
    v8h alo = *(const v8h*)(arow + k0 + kbA);
    v8h ahi = *(const v8h*)(arow + k0 + kbA + 8);
    v16h af = __builtin_shufflevector(alo, ahi, 0, 1, 2, 3, 4, 5, 6, 7,
                                      8, 9, 10, 11, 12, 13, 14, 15);
    v16h bf0 = *(const v16h*)(b0 + k0 + kbB);
    v16h bf1 = *(const v16h*)(b1 + k0 + kbB);
    v16h bf2 = *(const v16h*)(b2 + k0 + kbB);
    v16h bf3 = *(const v16h*)(b3 + k0 + kbB);
    acc0 = __builtin_amdgcn_wmma_f32_16x16x32_f16(false, af, false, bf0, (short)0, acc0, false, false);
    acc1 = __builtin_amdgcn_wmma_f32_16x16x32_f16(false, af, false, bf1, (short)0, acc1, false, false);
    acc2 = __builtin_amdgcn_wmma_f32_16x16x32_f16(false, af, false, bf2, (short)0, acc2, false, false);
    acc3 = __builtin_amdgcn_wmma_f32_16x16x32_f16(false, af, false, bf3, (short)0, acc3, false, false);
  }
  // C layout: VGPR r -> row (hi*8 + r), col = lane&15
  int row0 = tm + hi * 8;
#pragma unroll
  for (int r = 0; r < 8; ++r) {
    float* crow = C + (size_t)(row0 + r) * ldc;
    crow[tn0 + mr]      = acc0[r];
    crow[tn0 + 16 + mr] = acc1[r];
    crow[tn0 + 32 + mr] = acc2[r];
    crow[tn0 + 48 + mr] = acc3[r];
  }
}

// ---------- LayerNorm(+optional bias)+ReLU over rows, in place ----------
__global__ void ln_relu_rows(float* __restrict__ dat, const float* __restrict__ bias,
                             const float* __restrict__ g, const float* __restrict__ be,
                             int cols) {
  extern __shared__ float red[];
  int row = blockIdx.x, j = threadIdx.x;
  float x = dat[row * cols + j] + (bias ? bias[j] : 0.0f);
  float mean = block_reduce_sum(x, red) / (float)cols;
  float d = x - mean;
  float var = block_reduce_sum(d * d, red) / (float)cols;
  float y = d * rsqrtf(var + 1e-5f) * g[j] + be[j];
  dat[row * cols + j] = fmaxf(y, 0.0f);
}

// ---------- naive GEMM for tiny K matrices (thread per output) ----------
__global__ void gemm_naive(const float* __restrict__ A, int lda,
                           const float* __restrict__ Bm, int ldb,
                           const float* __restrict__ bias, float* __restrict__ C,
                           int M, int K, int N) {
  int idx = blockIdx.x * blockDim.x + threadIdx.x;
  if (idx >= M * N) return;
  int m = idx / N, n = idx % N;
  float acc = bias ? bias[n] : 0.0f;
  for (int k = 0; k < K; ++k) acc += A[m * lda + k] * Bm[k * ldb + n];
  C[idx] = acc;
}

// ---------- per-tick: v[j] = sum_n post[n,M-1]*sy_w1[4+n,j] + sy_b1[j] ----------
__global__ void syn_gemv(const float* __restrict__ post, const float* __restrict__ sy_w1,
                         const float* __restrict__ sy_b1, float* __restrict__ v) {
  int j = threadIdx.x;  // 256
  float acc = sy_b1[j];
  for (int n = 0; n < NN; ++n)
    acc += post[n * MM + (MM - 1)] * sy_w1[(4 + n) * 256 + j];
  v[j] = acc;
}

// ---------- h[b,:] = relu(ln(ca[b,:] + v)) ----------
__global__ void syn_hidden(const float* __restrict__ ca, const float* __restrict__ v,
                           const float* __restrict__ g, const float* __restrict__ be,
                           float* __restrict__ h) {
  __shared__ float red[256];
  int b = blockIdx.x, j = threadIdx.x;
  float x = ca[b * 256 + j] + v[j];
  float mean = block_reduce_sum(x, red) / 256.0f;
  float d = x - mean;
  float var = block_reduce_sum(d * d, red) / 256.0f;
  float y = d * rsqrtf(var + 1e-5f) * g[j] + be[j];
  h[b * 256 + j] = fmaxf(y, 0.0f);
}

// ---------- hbar[j] = mean_b h[b,j]  (deterministic column reduction) ----------
__global__ void col_mean_256(const float* __restrict__ h, float* __restrict__ hbar) {
  int j = threadIdx.x;  // 256
  float s = 0.0f;
  for (int b = 0; b < BB; ++b) s += h[b * 256 + j];
  hbar[j] = s * (1.0f / (float)BB);
}

// ---------- new_pre mean GEMV + shift pre history ----------
__global__ void pre_update(const float* __restrict__ hbar, const float* __restrict__ sy_w2,
                           const float* __restrict__ sy_b2, float* __restrict__ pre) {
  int n = blockIdx.x * blockDim.x + threadIdx.x;  // 4096
  float acc = sy_b2[n];
  for (int j = 0; j < 256; ++j) acc += hbar[j] * sy_w2[j * NN + n];
#pragma unroll
  for (int m = 0; m < MM - 1; ++m) pre[n * MM + m] = pre[n * MM + m + 1];
  pre[n * MM + MM - 1] = acc;
}

// ---------- neuron model: per-neuron 10x128 MAC + LN + dot, shift post ----------
__global__ void neuron_model(const float* __restrict__ pre, const float* __restrict__ nm_w1,
                             const float* __restrict__ nm_b1, const float* __restrict__ nm_g,
                             const float* __restrict__ nm_be, const float* __restrict__ nm_w2,
                             const float* __restrict__ nm_b2, float* __restrict__ post) {
  __shared__ float red[128];
  int n = blockIdx.x, h = threadIdx.x;  // 4096 x 128
  float pr[MM];
#pragma unroll
  for (int m = 0; m < MM; ++m) pr[m] = pre[n * MM + m];
  float z = nm_b1[n * 128 + h];
#pragma unroll
  for (int m = 0; m < MM; ++m) z += pr[m] * nm_w1[(n * MM + m) * 128 + h];
  float mean = block_reduce_sum(z, red) / 128.0f;
  float d = z - mean;
  float var = block_reduce_sum(d * d, red) / 128.0f;
  float y = fmaxf(d * rsqrtf(var + 1e-5f) * nm_g[n * 128 + h] + nm_be[n * 128 + h], 0.0f);
  float s = block_reduce_sum(y * nm_w2[n * 128 + h], red);
  if (h == 0) {
    float np = s + nm_b2[n];
#pragma unroll
    for (int m = 0; m < MM - 1; ++m) post[n * MM + m] = post[n * MM + m + 1];
    post[n * MM + MM - 1] = np;
  }
}

// ---------- final-tick sync over pairs ----------
__global__ void sync_pairs(const float* __restrict__ post, const int* __restrict__ pairs,
                           const float* __restrict__ decay, float* __restrict__ sv) {
  int p = blockIdx.x * blockDim.x + threadIdx.x;
  if (p >= PP) return;
  int i = pairs[2 * p], j = pairs[2 * p + 1];
  float dec = decay[p];
  float num = 0.0f, den = 0.0f;
#pragma unroll
  for (int k = 0; k < MM; ++k) {
    float r = expf(-dec * (float)(MM - 1 - k));
    num += post[i * MM + k] * r * post[j * MM + k];
    den += r;
  }
  sv[p] = num / (den + 1e-8f);
}

// ---------- u[j] = sum_p sv[p]*rd_w1[64+p, j] + rd_b1[j] ----------
__global__ void readout_gemv(const float* __restrict__ sv, const float* __restrict__ rd_w1,
                             const float* __restrict__ rd_b1, float* __restrict__ u) {
  int j = blockIdx.x * blockDim.x + threadIdx.x;  // 512
  float acc = rd_b1[j];
  for (int p = 0; p < PP; ++p) acc += sv[p] * rd_w1[(64 + p) * 512 + j];
  u[j] = acc;
}

// ---------- fused readout: LN+ReLU -> x rd_w2 -> LN+ReLU -> x rd_w3 ----------
__global__ void readout_final(const float* __restrict__ r1, const float* __restrict__ u,
                              const float* __restrict__ g1, const float* __restrict__ be1,
                              const float* __restrict__ rd_w2, const float* __restrict__ rd_b2,
                              const float* __restrict__ g2, const float* __restrict__ be2,
                              const float* __restrict__ rd_w3, const float* __restrict__ rd_b3,
                              float* __restrict__ out) {
  __shared__ float red[512];
  __shared__ float s1[512];
  __shared__ float r2s[64];
  int b = blockIdx.x, j = threadIdx.x;  // 1024 x 512
  float x = r1[b * 512 + j] + u[j];
  float mean = block_reduce_sum(x, red) / 512.0f;
  float d = x - mean;
  float var = block_reduce_sum(d * d, red) / 512.0f;
  s1[j] = fmaxf(d * rsqrtf(var + 1e-5f) * g1[j] + be1[j], 0.0f);
  __syncthreads();
  float r2raw = 0.0f;
  if (j < 64) {
    r2raw = rd_b2[j];
    for (int k = 0; k < 512; ++k) r2raw += s1[k] * rd_w2[k * 64 + j];
  }
  float m2 = block_reduce_sum((j < 64) ? r2raw : 0.0f, red) / 64.0f;
  float d2 = r2raw - m2;
  float v2 = block_reduce_sum((j < 64) ? d2 * d2 : 0.0f, red) / 64.0f;
  if (j < 64) r2s[j] = fmaxf(d2 * rsqrtf(v2 + 1e-5f) * g2[j] + be2[j], 0.0f);
  __syncthreads();
  if (j < 10) {
    float acc = rd_b3[j];
#pragma unroll
    for (int k = 0; k < 64; ++k) acc += r2s[k] * rd_w3[k * 10 + j];
    out[b * 10 + j] = acc;
  }
}

extern "C" void kernel_launch(void* const* d_in, const int* in_sizes, int n_in,
                              void* d_out, int out_size, void* d_ws, size_t ws_size,
                              hipStream_t stream) {
  (void)in_sizes; (void)n_in; (void)out_size; (void)ws_size;
  const float* x      = (const float*)d_in[0];
  const float* post0  = (const float*)d_in[1];
  const float* pre0   = (const float*)d_in[2];
  const int*   pairs  = (const int*)  d_in[3];
  const float* decay  = (const float*)d_in[4];
  const float* ie_w1  = (const float*)d_in[5];
  const float* ie_b1  = (const float*)d_in[6];
  const float* ie_g1  = (const float*)d_in[7];
  const float* ie_be1 = (const float*)d_in[8];
  const float* ie_w2  = (const float*)d_in[9];
  const float* ie_b2  = (const float*)d_in[10];
  const float* ie_g2  = (const float*)d_in[11];
  const float* ie_be2 = (const float*)d_in[12];
  const float* ie_w3  = (const float*)d_in[13];
  const float* ie_b3  = (const float*)d_in[14];
  const float* sy_w1  = (const float*)d_in[15];
  const float* sy_b1  = (const float*)d_in[16];
  const float* sy_g   = (const float*)d_in[17];
  const float* sy_be  = (const float*)d_in[18];
  const float* sy_w2  = (const float*)d_in[19];
  const float* sy_b2  = (const float*)d_in[20];
  const float* nm_w1  = (const float*)d_in[21];
  const float* nm_b1  = (const float*)d_in[22];
  const float* nm_g   = (const float*)d_in[23];
  const float* nm_be  = (const float*)d_in[24];
  const float* nm_w2  = (const float*)d_in[25];
  const float* nm_b2  = (const float*)d_in[26];
  // 27..28 key_w/key_b, 31..32 q_w/q_b: dead code (softmax over size-1 axis == 1)
  const float* val_w  = (const float*)d_in[29];
  const float* val_b  = (const float*)d_in[30];
  const float* rd_w1  = (const float*)d_in[33];
  const float* rd_b1  = (const float*)d_in[34];
  const float* rd_g1  = (const float*)d_in[35];
  const float* rd_be1 = (const float*)d_in[36];
  const float* rd_w2  = (const float*)d_in[37];
  const float* rd_b2  = (const float*)d_in[38];
  const float* rd_g2  = (const float*)d_in[39];
  const float* rd_be2 = (const float*)d_in[40];
  const float* rd_w3  = (const float*)d_in[41];
  const float* rd_b3  = (const float*)d_in[42];
  float* out = (float*)d_out;

  float* ws   = (float*)d_ws;
  float* g1   = ws; ws += BB * 128;   // encoder layer1 act
  float* e2   = ws; ws += BB * 32;    // encoder layer2 act
  float* enc  = ws; ws += BB * 4;     // enc
  float* vals = ws; ws += BB * 64;    // values (== att)
  float* post = ws; ws += NN * MM;    // mutable post history
  float* pre  = ws; ws += NN * MM;    // mutable pre history
  float* ca   = ws; ws += BB * 256;   // enc @ sy_w1[:4]
  float* hsyn = ws; ws += BB * 256;   // relu(ln(...)) per batch row
  float* vv   = ws; ws += 256;        // last_post @ sy_w1[4:] + sy_b1
  float* hbar = ws; ws += 256;        // mean_b hsyn
  float* sv   = ws; ws += PP;         // sync vector
  float* u    = ws; ws += 512;        // sv @ rd_w1[64:] + rd_b1
  float* r1   = ws; ws += BB * 512;   // values @ rd_w1[:64]
  // f16 staging (32B-aligned: preceding float region is a multiple of 32 bytes)
  _Float16* Ah = (_Float16*)ws;                 // up to 1024 x 800 halves
  _Float16* Bt = Ah + (size_t)BB * 800;         // up to 128 x 800 halves (K-major)

  // mutable histories (never touch d_in)
  copy_f32<<<(NN * MM + 255) / 256, 256, 0, stream>>>(post0, post, NN * MM);
  copy_f32<<<(NN * MM + 255) / 256, 256, 0, stream>>>(pre0, pre, NN * MM);

  // ---- encoder (once) ----
  // pack x(1024x784) -> Ah(1024x800 f16), ie_w1(784x128) -> Bt(128x800 f16, K-major)
  cvt_f16_pad  <<<(BB * 800 + 255) / 256, 256, 0, stream>>>(x, 784, 784, 800, Ah, BB * 800);
  cvt_f16_pad_t<<<(128 * 800 + 255) / 256, 256, 0, stream>>>(ie_w1, 784, 128, 800, Bt, 128 * 800);
  // g1 = x @ ie_w1   [WMMA: 64 row-strips x 2 wave-cols = 128 waves]
  wmma_gemm_packed<<<32, 128, 0, stream>>>(Ah, Bt, g1, 128, 800, 128 / 64);
  ln_relu_rows<<<BB, 128, 128 * sizeof(float), stream>>>(g1, ie_b1, ie_g1, ie_be1, 128);
  gemm_naive<<<(BB * 32 + 255) / 256, 256, 0, stream>>>(g1, 128, ie_w2, 32, ie_b2, e2, BB, 128, 32);
  ln_relu_rows<<<BB, 32, 32 * sizeof(float), stream>>>(e2, nullptr, ie_g2, ie_be2, 32);
  gemm_naive<<<(BB * 4 + 255) / 256, 256, 0, stream>>>(e2, 32, ie_w3, 4, ie_b3, enc, BB, 32, 4);
  gemm_naive<<<(BB * 64 + 255) / 256, 256, 0, stream>>>(enc, 4, val_w, 64, val_b, vals, BB, 4, 64);
  // ca = enc @ sy_w1[:4,:]  (batch-dependent part of syn_in @ sy_w1, tick-invariant)
  gemm_naive<<<(BB * 256 + 255) / 256, 256, 0, stream>>>(enc, 4, sy_w1, 256, nullptr, ca, BB, 4, 256);

  // ---- T ticks of recurrent dynamics ----
  for (int t = 0; t < TT; ++t) {
    syn_gemv<<<1, 256, 0, stream>>>(post, sy_w1, sy_b1, vv);
    syn_hidden<<<BB, 256, 0, stream>>>(ca, vv, sy_g, sy_be, hsyn);
    col_mean_256<<<1, 256, 0, stream>>>(hsyn, hbar);   // mean(h) -> mean(new_pre) by linearity
    pre_update<<<NN / 256, 256, 0, stream>>>(hbar, sy_w2, sy_b2, pre);
    neuron_model<<<NN, 128, 0, stream>>>(pre, nm_w1, nm_b1, nm_g, nm_be, nm_w2, nm_b2, post);
  }

  // ---- sync + readout (only the final tick's pred is returned) ----
  sync_pairs<<<PP / 256, 256, 0, stream>>>(post, pairs, decay, sv);
  readout_gemv<<<2, 256, 0, stream>>>(sv, rd_w1, rd_b1, u);
  // pack vals(1024x64) -> Ah, rd_w1[:64,:](64x512) -> Bt (K-major, Kpad=64)
  cvt_f16_pad  <<<(BB * 64 + 255) / 256, 256, 0, stream>>>(vals, 64, 64, 64, Ah, BB * 64);
  cvt_f16_pad_t<<<(512 * 64 + 255) / 256, 256, 0, stream>>>(rd_w1, 64, 512, 64, Bt, 512 * 64);
  // r1 = vals @ rd_w1[:64]   [WMMA: 64 row-strips x 8 wave-cols = 512 waves]
  wmma_gemm_packed<<<128, 128, 0, stream>>>(Ah, Bt, r1, 512, 64, 512 / 64);
  readout_final<<<BB, 512, 0, stream>>>(r1, u, rd_g1, rd_be1, rd_w2, rd_b2,
                                        rd_g2, rd_be2, rd_w3, rd_b3, out);
}
